// DFConv_57543971832540
// MI455X (gfx1250) — compile-verified, matrix operand order checked by reading, not run on hardware
//
#include <hip/hip_runtime.h>
#include <hip/hip_bf16.h>

// ---------------- problem constants ----------------
constexpr int Bn   = 8;
constexpr int C1   = 256;
constexpr int C2   = 256;
constexpr int Hh   = 64;
constexpr int Ww   = 64;
constexpr int HW   = Hh * Ww;              // 4096
constexpr int KK   = 9;                    // 3x3 taps
constexpr int Kdim = C1 * KK;              // 2304 (GEMM K)
constexpr int Mrows = Bn * HW;             // 32768 (GEMM M)
constexpr int Noff = 18;                   // offset channels
constexpr int Nmsk = 9;                    // mask channels
constexpr int N1   = 32;                   // 27 padded to 32
constexpr float MAXOFF = 16.0f;            // max(H,W)/4

typedef __attribute__((ext_vector_type(16))) __bf16 v16bf;
typedef __attribute__((ext_vector_type(8)))  float  v8f;

union Frag {
    v16bf v;
    uint4 q[2];
};

__device__ __forceinline__ unsigned short f2bf(float f) {
    unsigned int u = __float_as_uint(f);
    u += 0x7FFFu + ((u >> 16) & 1u);       // round-to-nearest-even
    return (unsigned short)(u >> 16);
}

// Load one 16x32 bf16 fragment (A from [M][K] row-major, or B from [N][K] row-major).
// Lane l: row = base_row + (l&15), half h = l>>4.
// VGPRs 0..3 <- K = k0 + 8h .. k0 + 8h + 7     (one b128)
// VGPRs 4..7 <- K = k0 + 16 + 8h .. + 23       (one b128)
__device__ __forceinline__ void load_frag(const unsigned short* __restrict__ base,
                                          int row, int k0, int h, Frag& f) {
    const unsigned short* p = base + (size_t)row * Kdim + k0 + (h << 3);
    f.q[0] = *(const uint4*)(p);
    f.q[1] = *(const uint4*)(p + 16);
}

// ---------------- kernel 1: pack weights to bf16 ----------------
// w1[32][2304]: rows 0..17 = offset_w, 18..26 = mod_w, 27..31 = 0
// wT[256][2304]: reg_w flattened (already [o][c*9+kk])
__global__ void pack_weights_kernel(const float* __restrict__ offset_w,
                                    const float* __restrict__ mod_w,
                                    const float* __restrict__ reg_w,
                                    unsigned short* __restrict__ w1,
                                    unsigned short* __restrict__ wT) {
    int i = blockIdx.x * blockDim.x + threadIdx.x;
    const int n1 = N1 * Kdim;                  // 73728
    const int n2 = C2 * Kdim;                  // 589824
    if (i < n1) {
        int n = i / Kdim;
        int k = i - n * Kdim;
        float v = 0.0f;
        if (n < Noff)            v = offset_w[n * Kdim + k];
        else if (n < Noff + Nmsk) v = mod_w[(n - Noff) * Kdim + k];
        w1[i] = f2bf(v);
    } else if (i < n1 + n2) {
        int j = i - n1;
        wT[j] = f2bf(reg_w[j]);
    }
}

// ---------------- kernel 2: im2col of x into bf16 [M][K] ----------------
// grid (Kdim/256, Mrows), block 256; thread -> one (m,k) element
__global__ void im2col_kernel(const float* __restrict__ x,
                              unsigned short* __restrict__ xcol) {
    int k = blockIdx.x * blockDim.x + threadIdx.x;   // 0..2303
    int m = blockIdx.y;                              // 0..32767
    int c  = k / 9;
    int t  = k - 9 * c;
    int ky = t / 3;
    int kx = t - 3 * ky;
    int b  = m >> 12;
    int hw = m & 4095;
    int y  = hw >> 6;
    int xx = hw & 63;
    int yy = y + ky - 1;
    int xs = xx + kx - 1;
    float v = 0.0f;
    if ((unsigned)yy < (unsigned)Hh && (unsigned)xs < (unsigned)Ww)
        v = x[(((size_t)(b * C1 + c)) << 12) + (yy << 6) + xs];
    xcol[(size_t)m * Kdim + k] = f2bf(v);
}

// ---------------- kernel 3: WMMA GEMM N=32 with clip/sigmoid epilogue ----------
// block = 128 threads = 4 waves; each wave: 64 rows x 32 cols; grid.x = 128
__global__ __launch_bounds__(128) void conv1_wmma_kernel(
        const unsigned short* __restrict__ xcol,
        const unsigned short* __restrict__ w1,
        const float* __restrict__ offset_b,
        const float* __restrict__ mod_b,
        float* __restrict__ off_out,   // [B][18][HW]
        float* __restrict__ msk_out) { // [B][9][HW]
    const int lane = threadIdx.x & 31;
    const int wave = threadIdx.x >> 5;
    const int m0   = blockIdx.x * 256 + wave * 64;
    const int rsel = lane & 15;
    const int h    = lane >> 4;

    v8f zero = {};
    v8f acc[4][2];
#pragma unroll
    for (int i = 0; i < 4; ++i)
#pragma unroll
        for (int j = 0; j < 2; ++j) acc[i][j] = zero;

    for (int k0 = 0; k0 < Kdim; k0 += 32) {
        Frag a[4], bq[2];
#pragma unroll
        for (int i = 0; i < 4; ++i) load_frag(xcol, m0 + i * 16 + rsel, k0, h, a[i]);
#pragma unroll
        for (int j = 0; j < 2; ++j) load_frag(w1, j * 16 + rsel, k0, h, bq[j]);
#pragma unroll
        for (int i = 0; i < 4; ++i)
#pragma unroll
            for (int j = 0; j < 2; ++j)
                acc[i][j] = __builtin_amdgcn_wmma_f32_16x16x32_bf16(
                    false, a[i].v, false, bq[j].v, (short)0, acc[i][j], false, false);
    }

    // epilogue: D element (vgpr r, lane l) = (M = r + 8h, N = l&15)
#pragma unroll
    for (int j = 0; j < 2; ++j) {
        int n = j * 16 + rsel;
#pragma unroll
        for (int i = 0; i < 4; ++i) {
#pragma unroll
            for (int r = 0; r < 8; ++r) {
                int m  = m0 + i * 16 + r + 8 * h;
                int b  = m >> 12;
                int hw = m & 4095;
                float v = acc[i][j][r];
                if (n < Noff) {
                    v += offset_b[n];
                    v = fminf(fmaxf(v, -MAXOFF), MAXOFF);
                    off_out[b * (Noff * HW) + n * HW + hw] = v;
                } else if (n < Noff + Nmsk) {
                    v += mod_b[n - Noff];
                    msk_out[b * (Nmsk * HW) + (n - Noff) * HW + hw] =
                        1.0f / (1.0f + __expf(-v));
                }
            }
        }
    }
}

// ---------------- kernel 4: bilinear sampling + modulation -> val bf16 [M][K] --
// grid (Kdim/256, Mrows), block 256
__global__ void sample_kernel(const float* __restrict__ x,
                              const float* __restrict__ off,   // [B][18][HW]
                              const float* __restrict__ msk,   // [B][9][HW]
                              unsigned short* __restrict__ val) {
    int k = blockIdx.x * blockDim.x + threadIdx.x;   // 0..2303
    int m = blockIdx.y;                              // 0..32767
    int c  = k / 9;
    int t  = k - 9 * c;          // tap index kk
    int ky = t / 3;
    int kx = t - 3 * ky;
    int b  = m >> 12;
    int hw = m & 4095;
    int y  = hw >> 6;
    int xx = hw & 63;

    const float* offB = off + b * (Noff * HW);
    float dy = offB[(2 * t)     * HW + hw];
    float dx = offB[(2 * t + 1) * HW + hw];
    float mk = msk[b * (Nmsk * HW) + t * HW + hw];

    float py = dy + (float)(y + ky - 1);
    float px = dx + (float)(xx + kx - 1);
    float y0f = floorf(py), x0f = floorf(px);
    float ly = py - y0f, lx = px - x0f;
    int y0 = (int)y0f, x0 = (int)x0f;

    const float* xb = x + (((size_t)(b * C1 + c)) << 12);
    float v00 = 0.f, v01 = 0.f, v10 = 0.f, v11 = 0.f;
    if ((unsigned)y0       < 64u && (unsigned)x0       < 64u) v00 = xb[(y0 << 6) + x0];
    if ((unsigned)y0       < 64u && (unsigned)(x0 + 1) < 64u) v01 = xb[(y0 << 6) + x0 + 1];
    if ((unsigned)(y0 + 1) < 64u && (unsigned)x0       < 64u) v10 = xb[((y0 + 1) << 6) + x0];
    if ((unsigned)(y0 + 1) < 64u && (unsigned)(x0 + 1) < 64u) v11 = xb[((y0 + 1) << 6) + x0 + 1];

    float v = (1.f - ly) * ((1.f - lx) * v00 + lx * v01)
            +        ly  * ((1.f - lx) * v10 + lx * v11);
    val[(size_t)m * Kdim + k] = f2bf(v * mk);
}

// ---------------- kernel 5: main WMMA GEMM M=32768 N=256 K=2304 ----------------
// block = 128 threads = 4 waves; each wave owns a 64x64 output tile.
// grid = (M/256, N/64) = (128, 4)
__global__ __launch_bounds__(128) void gemm_wmma_kernel(
        const unsigned short* __restrict__ val,  // [M][K] bf16
        const unsigned short* __restrict__ wT,   // [N][K] bf16
        float* __restrict__ out) {               // [B][C2][HW] fp32
    const int lane = threadIdx.x & 31;
    const int wave = threadIdx.x >> 5;
    const int m0   = blockIdx.x * 256 + wave * 64;
    const int n0   = blockIdx.y * 64;
    const int rsel = lane & 15;
    const int h    = lane >> 4;

    v8f zero = {};
    v8f acc[4][4];
#pragma unroll
    for (int i = 0; i < 4; ++i)
#pragma unroll
        for (int j = 0; j < 4; ++j) acc[i][j] = zero;

    for (int k0 = 0; k0 < Kdim; k0 += 32) {
        Frag a[4], bq[4];
#pragma unroll
        for (int i = 0; i < 4; ++i) load_frag(val, m0 + i * 16 + rsel, k0, h, a[i]);
#pragma unroll
        for (int j = 0; j < 4; ++j) load_frag(wT, n0 + j * 16 + rsel, k0, h, bq[j]);
#pragma unroll
        for (int i = 0; i < 4; ++i)
#pragma unroll
            for (int j = 0; j < 4; ++j)
                acc[i][j] = __builtin_amdgcn_wmma_f32_16x16x32_bf16(
                    false, a[i].v, false, bq[j].v, (short)0, acc[i][j], false, false);
    }

#pragma unroll
    for (int i = 0; i < 4; ++i) {
#pragma unroll
        for (int j = 0; j < 4; ++j) {
            int n = n0 + j * 16 + rsel;
#pragma unroll
            for (int r = 0; r < 8; ++r) {
                int m  = m0 + i * 16 + r + 8 * h;
                int b  = m >> 12;
                int hw = m & 4095;
                out[(size_t)b * (C2 * HW) + (size_t)n * HW + hw] = acc[i][j][r];
            }
        }
    }
}

// ---------------- host launcher ----------------
extern "C" void kernel_launch(void* const* d_in, const int* in_sizes, int n_in,
                              void* d_out, int out_size, void* d_ws, size_t ws_size,
                              hipStream_t stream) {
    const float* x        = (const float*)d_in[0];
    const float* offset_w = (const float*)d_in[1];
    const float* offset_b = (const float*)d_in[2];
    const float* mod_w    = (const float*)d_in[3];
    const float* mod_b    = (const float*)d_in[4];
    const float* reg_w    = (const float*)d_in[5];
    float* out = (float*)d_out;

    // workspace partition (~156 MB total)
    char* ws = (char*)d_ws;
    size_t off_col = 0;
    size_t sz_col  = (size_t)Mrows * Kdim * sizeof(unsigned short);  // 150,994,944
    size_t off_w1  = off_col + sz_col;
    size_t sz_w1   = (size_t)N1 * Kdim * sizeof(unsigned short);     // 147,456
    size_t off_wT  = off_w1 + sz_w1;
    size_t sz_wT   = (size_t)C2 * Kdim * sizeof(unsigned short);     // 1,179,648
    size_t off_off = off_wT + sz_wT;
    size_t sz_off  = (size_t)Bn * Noff * HW * sizeof(float);         // 2,359,296
    size_t off_msk = off_off + sz_off;

    unsigned short* colbuf = (unsigned short*)(ws + off_col);  // im2col, then reused for val
    unsigned short* w1     = (unsigned short*)(ws + off_w1);
    unsigned short* wT     = (unsigned short*)(ws + off_wT);
    float*          offbuf = (float*)(ws + off_off);
    float*          mskbuf = (float*)(ws + off_msk);

    // 1. pack weights to bf16
    {
        int total = N1 * Kdim + C2 * Kdim;
        pack_weights_kernel<<<(total + 255) / 256, 256, 0, stream>>>(
            offset_w, mod_w, reg_w, w1, wT);
    }
    // 2. im2col x -> bf16 [M][K]
    im2col_kernel<<<dim3(Kdim / 256, Mrows), 256, 0, stream>>>(x, colbuf);
    // 3. offset/mask conv via WMMA, fused clip + sigmoid
    conv1_wmma_kernel<<<dim3(Mrows / 256, 1), 128, 0, stream>>>(
        colbuf, w1, offset_b, mod_b, offbuf, mskbuf);
    // 4. deformable bilinear sampling + modulation -> val bf16 (overwrites colbuf)
    sample_kernel<<<dim3(Kdim / 256, Mrows), 256, 0, stream>>>(
        x, offbuf, mskbuf, colbuf);
    // 5. main GEMM via WMMA -> NCHW fp32 output
    gemm_wmma_kernel<<<dim3(Mrows / 256, C2 / 64), 128, 0, stream>>>(
        colbuf, wT, out);
}
